// KmeansVectorQuantizer_42949673163
// MI455X (gfx1250) — compile-verified, matrix operand order checked by reading, not run on hardware
//
#include <hip/hip_runtime.h>
#include <hip/hip_bf16.h>

// ---------------- problem constants ----------------
#define B_ 16
#define T_ 2048
#define C_ 512
#define G_ 2
#define V_ 320
#define D_ 256
#define TM 32                       // t-rows per block (2 row-tiles per wave)
#define TILES_T (T_/TM)             // 64 t-tiles
#define NBLK (B_*G_*TILES_T)        // 2048 blocks for stats/main
#define NTOT (B_*T_*C_)             // 16777216 output elements of x_out

// ---------------- vector types ----------------
typedef __attribute__((ext_vector_type(16))) __bf16 v16bf;
typedef __attribute__((ext_vector_type(8)))  __bf16 bf16x8;
typedef __attribute__((ext_vector_type(8)))  float  v8f;
typedef __attribute__((ext_vector_type(4)))  float  f32x4;

#define LDS_ROW 264   // bf16 elems per LDS row: 256 + 8 pad (16B) -> conflict-free reads
#define ZROW    257   // f32 elems per ze row

union BFrag { v16bf v; bf16x8 h[2]; };

__device__ __forceinline__ v8f zero8() {
  v8f z; for (int i = 0; i < 8; ++i) z[i] = 0.f; return z;
}

// A fragment (16x32 bf16, MxK): lane m=lane&15, half=lane>>4.
// elems 0..7  = A[m][kb + half*8 .. +7], elems 8..15 = A[m][kb+16+half*8 .. +7]
__device__ __forceinline__ v16bf ldsAFrag(const __bf16* lds, int m, int half, int kb) {
  BFrag f;
  f.h[0] = *reinterpret_cast<const bf16x8*>(lds + m * LDS_ROW + kb + half * 8);
  f.h[1] = *reinterpret_cast<const bf16x8*>(lds + m * LDS_ROW + kb + 16 + half * 8);
  return f.v;
}

// B fragment (32x16 bf16, KxN): lane holds column n; B[k][n] = Mrow[n][k] with
// Mrow K-contiguous in memory (NT gemm), so two contiguous 16B loads per lane.
__device__ __forceinline__ v16bf glbBFrag(const __bf16* rowptr, int half, int kb) {
  BFrag f;
  f.h[0] = *reinterpret_cast<const bf16x8*>(rowptr + kb + half * 8);
  f.h[1] = *reinterpret_cast<const bf16x8*>(rowptr + kb + 16 + half * 8);
  return f.v;
}

// Cooperative load of a TM x 256 f32 x-tile -> bf16 LDS tile (128 threads)
__device__ __forceinline__ void loadXTile(const float* xbase, __bf16* abf, int tid) {
  for (int p = 0; p < TM / 16; ++p) {
    int row = p * 16 + (tid >> 3);     // 0..TM-1
    int c0  = (tid & 7) << 5;          // 0..224 step 32
    const float* xp = xbase + (size_t)row * C_ + c0;
    __bf16* dst = abf + row * LDS_ROW + c0;
    for (int u = 0; u < 32; u += 8) {
      f32x4 a = *reinterpret_cast<const f32x4*>(xp + u);
      f32x4 b = *reinterpret_cast<const f32x4*>(xp + u + 4);
      bf16x8 o;
      o[0] = (__bf16)a[0]; o[1] = (__bf16)a[1]; o[2] = (__bf16)a[2]; o[3] = (__bf16)a[3];
      o[4] = (__bf16)b[0]; o[5] = (__bf16)b[1]; o[6] = (__bf16)b[2]; o[7] = (__bf16)b[3];
      *reinterpret_cast<bf16x8*>(dst + u) = o;
    }
  }
}

// Per-wave GEMM1: 32(t) x 64(o) tile of h = Xtile * W^T, bf16 WMMA, f32 accum.
// Each B fragment feeds two WMMAs (row-tiles 0 and 1) -> halved L2 B-traffic.
__device__ __forceinline__ void gemm1(const __bf16* abf, const __bf16* Wbf,
                                      int w, int lane, v8f acc[2][4]) {
  int m = lane & 15, half = lane >> 4;
  for (int rt = 0; rt < 2; ++rt)
    for (int j = 0; j < 4; ++j) acc[rt][j] = zero8();
  for (int kk = 0; kk < 8; ++kk) {
    int kb = kk * 32;
    v16bf a0 = ldsAFrag(abf, m, half, kb);
    v16bf a1 = ldsAFrag(abf + 16 * LDS_ROW, m, half, kb);
    for (int j = 0; j < 4; ++j) {
      int o = (w * 4 + j) * 16 + m;                 // n == lane&15
      v16bf bm = glbBFrag(Wbf + (size_t)o * D_, half, kb);
      acc[0][j] = __builtin_amdgcn_wmma_f32_16x16x32_bf16(
          false, a0, false, bm, (short)0, acc[0][j], false, false);
      acc[1][j] = __builtin_amdgcn_wmma_f32_16x16x32_bf16(
          false, a1, false, bm, (short)0, acc[1][j], false, false);
    }
  }
}

// ---------------- setup kernels ----------------
__global__ void vq_convert_kernel(const float* __restrict__ wsrc,
                                  const float* __restrict__ esrc,
                                  __bf16* __restrict__ Wbf,
                                  __bf16* __restrict__ Ebf,
                                  float* __restrict__ counts) {
  int gid = blockIdx.x * blockDim.x + threadIdx.x;
  const int NW = C_ * D_;           // 131072
  const int NE = V_ * G_ * D_;      // 163840
  for (int i = gid; i < NW + NE; i += gridDim.x * blockDim.x) {
    if (i < NW) Wbf[i] = (__bf16)wsrc[i];
    else        Ebf[i - NW] = (__bf16)esrc[i - NW];
  }
  if (gid < V_ * G_) counts[gid] = 0.f;   // reset every call (deterministic)
}

__global__ void __launch_bounds__(32) vq_esq_kernel(const float* __restrict__ esrc,
                                                    float* __restrict__ esq) {
  int vg = blockIdx.x;              // v*G + g
  int lane = threadIdx.x;
  const float* p = esrc + (size_t)vg * D_;
  float s = 0.f;
  for (int k = 0; k < 8; ++k) { float v = p[lane + 32 * k]; s += v * v; }
  for (int off = 16; off > 0; off >>= 1) s += __shfl_xor(s, off, 32);
  if (lane == 0) esq[vg] = s;
}

// ---------------- pass 1: GroupNorm statistics ----------------
__global__ void __launch_bounds__(128) vq_stats_kernel(const float* __restrict__ x,
                                                       const __bf16* __restrict__ Wbf,
                                                       float* __restrict__ statsPartial) {
  __shared__ __align__(16) __bf16 abf[TM * LDS_ROW];
  __shared__ float sbuf[8];
  int tid = threadIdx.x, bid = blockIdx.x;
  int tt = bid & (TILES_T - 1), bg = bid >> 6;
  int b = bg >> 1, g = bg & 1;
  int t0 = tt * TM;
  const float* xbase = x + ((size_t)(b * T_ + t0)) * C_ + g * D_;
  loadXTile(xbase, abf, tid);
  __syncthreads();
  int w = tid >> 5, lane = tid & 31;
  v8f acc[2][4];
  gemm1(abf, Wbf, w, lane, acc);
  float s = 0.f, sq = 0.f;
  for (int rt = 0; rt < 2; ++rt)
    for (int j = 0; j < 4; ++j)
      for (int r = 0; r < 8; ++r) { float h = acc[rt][j][r]; s += h; sq += h * h; }
  for (int off = 16; off > 0; off >>= 1) {
    s  += __shfl_xor(s,  off, 32);
    sq += __shfl_xor(sq, off, 32);
  }
  if (lane == 0) { sbuf[w] = s; sbuf[4 + w] = sq; }
  __syncthreads();
  if (tid == 0) {
    statsPartial[bid * 2 + 0] = sbuf[0] + sbuf[1] + sbuf[2] + sbuf[3];
    statsPartial[bid * 2 + 1] = sbuf[4] + sbuf[5] + sbuf[6] + sbuf[7];
  }
}

__global__ void __launch_bounds__(64) vq_stats_final(const float* __restrict__ statsPartial,
                                                     float* __restrict__ muArr,
                                                     float* __restrict__ rstdArr) {
  __shared__ float rs[64], rq[64];
  int bg = blockIdx.x, tid = threadIdx.x;
  int q = bg * TILES_T + tid;
  rs[tid] = statsPartial[q * 2 + 0];
  rq[tid] = statsPartial[q * 2 + 1];
  __syncthreads();
  for (int s = 32; s > 0; s >>= 1) {
    if (tid < s) { rs[tid] += rs[tid + s]; rq[tid] += rq[tid + s]; }
    __syncthreads();
  }
  if (tid == 0) {
    const float inv = 1.f / ((float)D_ * (float)T_);
    float mean = rs[0] * inv;
    float var  = rq[0] * inv - mean * mean;
    muArr[bg] = mean;
    rstdArr[bg] = rsqrtf(var + 1e-5f);
  }
}

// ---------------- pass 2: recompute h, normalize, VQ, outputs ----------------
__global__ void __launch_bounds__(128) vq_main_kernel(
    const float* __restrict__ x, const __bf16* __restrict__ Wbf,
    const __bf16* __restrict__ Ebf, const float* __restrict__ emb,
    const float* __restrict__ gn_w, const float* __restrict__ gn_b,
    const float* __restrict__ muArr, const float* __restrict__ rstdArr,
    const float* __restrict__ esq, float* __restrict__ counts,
    float* __restrict__ lossPartial, float* __restrict__ out) {
  __shared__ __align__(16) __bf16 abf[TM * LDS_ROW];  // x tile, then ze (bf16)
  __shared__ float zef[TM * ZROW];                    // ze (f32) for loss
  __shared__ float partScore[TM * 64];
  __shared__ int   partV[TM * 64];
  __shared__ int   idxSh[TM];
  __shared__ float red[128];

  int tid = threadIdx.x, bid = blockIdx.x;
  int tt = bid & (TILES_T - 1), bg = bid >> 6;
  int b = bg >> 1, g = bg & 1;
  int t0 = tt * TM;
  const float* xbase = x + ((size_t)(b * T_ + t0)) * C_ + g * D_;
  loadXTile(xbase, abf, tid);
  __syncthreads();
  int w = tid >> 5, lane = tid & 31;
  int m = lane & 15, half = lane >> 4;

  v8f acc[2][4];
  gemm1(abf, Wbf, w, lane, acc);
  __syncthreads();                 // all x-tile reads done before ze overwrite

  float mu = muArr[bg], rstd = rstdArr[bg];
  for (int rt = 0; rt < 2; ++rt) {
    for (int j = 0; j < 4; ++j) {
      int o = (w * 4 + j) * 16 + m;
      float gw = gn_w[g * D_ + o], gb = gn_b[g * D_ + o];
      for (int r = 0; r < 8; ++r) {
        int t = rt * 16 + r + half * 8;
        float zv = (acc[rt][j][r] - mu) * rstd * gw + gb;
        zef[t * ZROW + o] = zv;
        abf[t * LDS_ROW + o] = (__bf16)zv;
      }
    }
  }
  __syncthreads();

  // GEMM2: dist score = esq[v] - 2*ze.E_v  (|ze|^2 constant per row -> dropped)
  const __bf16* Eg = Ebf + g * D_;   // E row v at Eg + v*(G*D), K contiguous
  float bs[2][8]; int bv[2][8];
  for (int rt = 0; rt < 2; ++rt)
    for (int r = 0; r < 8; ++r) { bs[rt][r] = 3.402823466e38f; bv[rt][r] = 0; }
  for (int s5 = 0; s5 < 5; ++s5) {
    int vbase = (w * 5 + s5) * 16;
    v8f a2[2]; a2[0] = zero8(); a2[1] = zero8();
    for (int kk = 0; kk < 8; ++kk) {
      int kb = kk * 32;
      v16bf a0 = ldsAFrag(abf, m, half, kb);
      v16bf a1 = ldsAFrag(abf + 16 * LDS_ROW, m, half, kb);
      v16bf bm = glbBFrag(Eg + (size_t)(vbase + m) * (G_ * D_), half, kb);
      a2[0] = __builtin_amdgcn_wmma_f32_16x16x32_bf16(
          false, a0, false, bm, (short)0, a2[0], false, false);
      a2[1] = __builtin_amdgcn_wmma_f32_16x16x32_bf16(
          false, a1, false, bm, (short)0, a2[1], false, false);
    }
    int vv = vbase + m;
    float e2 = esq[vv * G_ + g];
    for (int rt = 0; rt < 2; ++rt) {
      for (int r = 0; r < 8; ++r) {
        float sc = e2 - 2.f * a2[rt][r];
        if (sc < bs[rt][r] || (sc == bs[rt][r] && vv < bv[rt][r])) {
          bs[rt][r] = sc; bv[rt][r] = vv;
        }
      }
    }
  }
  for (int rt = 0; rt < 2; ++rt) {
    for (int r = 0; r < 8; ++r) {
      int t = rt * 16 + r + half * 8;
      partScore[t * 64 + w * 16 + m] = bs[rt][r];
      partV[t * 64 + w * 16 + m]     = bv[rt][r];
    }
  }
  __syncthreads();
  if (tid < TM) {
    float best = 3.402823466e38f; int bvv = 0;
    for (int s = 0; s < 64; ++s) {
      float sc = partScore[tid * 64 + s]; int vv = partV[tid * 64 + s];
      if (sc < best || (sc == best && vv < bvv)) { best = sc; bvv = vv; }
    }
    idxSh[tid] = bvv;
  }
  __syncthreads();

  // outputs: x_out rows + per-block loss partial + counts
  float lacc = 0.f;
  for (int t = 0; t < TM; ++t) {
    int idx = idxSh[t];
    const float* ev = emb + ((size_t)idx * G_ + g) * D_;
    float* op = out + ((size_t)(b * T_ + t0 + t)) * C_ + g * D_;
    for (int o = tid; o < D_; o += 128) {
      float e = ev[o];
      float d = e - zef[t * ZROW + o];
      lacc += d * d;
      op[o] = e;
    }
  }
  red[tid] = lacc;
  __syncthreads();
  for (int s = 64; s > 0; s >>= 1) {
    if (tid < s) red[tid] += red[tid + s];
    __syncthreads();
  }
  if (tid == 0) lossPartial[bid] = red[0];
  if (tid < TM) atomicAdd(&counts[idxSh[tid] * G_ + g], 1.0f);  // integer-exact f32
}

// ---------------- final scalars ----------------
__global__ void __launch_bounds__(256) vq_final_kernel(const float* __restrict__ lossPartial,
                                                       const float* __restrict__ counts,
                                                       float* __restrict__ out) {
  __shared__ float r0[256], r1[256], r2[256];
  int tid = threadIdx.x;
  float ls = 0.f;
  for (int i = tid; i < NBLK; i += 256) ls += lossPartial[i];
  float h0 = 0.f, h1 = 0.f;
  const float invN = 1.f / (float)(B_ * T_);
  for (int v = tid; v < V_; v += 256) {
    float p0 = counts[v * G_ + 0] * invN;
    float p1 = counts[v * G_ + 1] * invN;
    h0 += p0 * logf(p0 + 1e-7f);
    h1 += p1 * logf(p1 + 1e-7f);
  }
  r0[tid] = ls; r1[tid] = h0; r2[tid] = h1;
  __syncthreads();
  for (int s = 128; s > 0; s >>= 1) {
    if (tid < s) { r0[tid] += r0[tid + s]; r1[tid] += r1[tid + s]; r2[tid] += r2[tid + s]; }
    __syncthreads();
  }
  if (tid == 0) {
    out[NTOT]     = r0[0] * (1.25f / (float)NTOT);     // latent + GAMMA*commitment
    out[NTOT + 1] = expf(-r1[0]) + expf(-r2[0]);       // code perplexity
  }
}

// ---------------- launcher ----------------
extern "C" void kernel_launch(void* const* d_in, const int* in_sizes, int n_in,
                              void* d_out, int out_size, void* d_ws, size_t ws_size,
                              hipStream_t stream) {
  const float* x     = (const float*)d_in[0];   // (B,T,C)
  const float* convw = (const float*)d_in[1];   // (C,D)
  const float* gnw   = (const float*)d_in[2];   // (C,)
  const float* gnb   = (const float*)d_in[3];   // (C,)
  const float* emb   = (const float*)d_in[4];   // (V,G,D)
  float* out = (float*)d_out;

  // workspace layout (~630 KB total)
  float* wsf          = (float*)d_ws;
  float* counts       = wsf;                        // 640
  float* muArr        = counts + 640;               // 32
  float* rstdArr      = muArr + 32;                 // 32
  float* esq          = rstdArr + 32;               // 640
  float* lossPartial  = esq + 640;                  // 2048
  float* statsPartial = lossPartial + 2048;         // 4096
  __bf16* Wbf = (__bf16*)(statsPartial + 4096);     // 131072 bf16 (16B aligned)
  __bf16* Ebf = Wbf + (size_t)C_ * D_;              // 163840 bf16

  vq_convert_kernel<<<576, 512, 0, stream>>>(convw, emb, Wbf, Ebf, counts);
  vq_esq_kernel<<<V_ * G_, 32, 0, stream>>>(emb, esq);
  vq_stats_kernel<<<NBLK, 128, 0, stream>>>(x, Wbf, statsPartial);
  vq_stats_final<<<B_ * G_, 64, 0, stream>>>(statsPartial, muArr, rstdArr);
  vq_main_kernel<<<NBLK, 128, 0, stream>>>(x, Wbf, Ebf, emb, gnw, gnb,
                                           muArr, rstdArr, esq, counts,
                                           lossPartial, out);
  vq_final_kernel<<<1, 256, 0, stream>>>(lossPartial, counts, out);
}
